// Quantizer_31044023615534
// MI455X (gfx1250) — matrix operand order repaired and checked by reading, not compiled, not run on hardware
//
#include <hip/hip_runtime.h>

// ---------------------------------------------------------------------------
// VQ quantizer forward for MI455X (gfx1250): distance GEMM via
// v_wmma_f32_16x16x32_f16, LDS-resident f16 codebook, per-row argmin,
// NCHW gather-write, histogram + perplexity.
// ---------------------------------------------------------------------------

typedef _Float16 half16v __attribute__((ext_vector_type(16)));
typedef _Float16 half8v  __attribute__((ext_vector_type(8)));
typedef float    float8v __attribute__((ext_vector_type(8)));

#define KCODES   1024
#define DDIM     64
#define NROWS    131072          // B*H*W = 32*64*64
#define HW       4096            // H*W
#define CHW      (64 * 4096)     // C*H*W
#define LDS_ROW  72              // padded halves per codebook row (bank-conflict-free)
#define WAVES    8               // waves per block (wave32)

#define CB_LDS_BYTES (KCODES * LDS_ROW * 2)             // 147456
#define HN_LDS_BYTES (KCODES * 4)                       // 4096
#define SMEM_BYTES   (CB_LDS_BYTES + HN_LDS_BYTES + WAVES * 16 * 4)

// workspace layout (bytes)
#define WS_HN_OFF   (KCODES * DDIM * 2)                 // after f16 codebook
#define WS_CNT_OFF  (WS_HN_OFF + KCODES * 4)

// ---------------------------------------------------------------------------
// Kernel 1: codebook f32 -> f16, and 0.5*||c_k||^2. One wave per code row.
// ---------------------------------------------------------------------------
__global__ void __launch_bounds__(256) vq_prep(const float* __restrict__ cb,
                                               _Float16* __restrict__ cbh,
                                               float* __restrict__ hn) {
    const int wave = threadIdx.x >> 5;
    const int lane = threadIdx.x & 31;
    const int k = blockIdx.x * WAVES + wave;           // 128 blocks * 8 = 1024 rows
    const float* row = cb + (size_t)k * DDIM;
    float c0 = row[lane];
    float c1 = row[lane + 32];
    cbh[(size_t)k * DDIM + lane]      = (_Float16)c0;
    cbh[(size_t)k * DDIM + lane + 32] = (_Float16)c1;
    float s = c0 * c0 + c1 * c1;
    #pragma unroll
    for (int m = 16; m >= 1; m >>= 1) s += __shfl_xor(s, m, 32);
    if (lane == 0) hn[k] = 0.5f * s;
}

// ---------------------------------------------------------------------------
// Kernel 2: main. Each wave: 16 rows of X vs all 1024 codes via WMMA.
// Score s(m,n) = x_m . c_n - 0.5||c_n||^2  (accumulator preloaded with -hn).
// argmax over n == argmin of squared distance.
// ---------------------------------------------------------------------------
__global__ void __launch_bounds__(256) vq_main(const float* __restrict__ X,
                                               const float* __restrict__ CB,
                                               const _Float16* __restrict__ CBH,
                                               const float* __restrict__ HN,
                                               unsigned* __restrict__ counts,
                                               float* __restrict__ out) {
    extern __shared__ char smem[];
    _Float16* cbL = (_Float16*)smem;
    float*    hnL = (float*)(smem + CB_LDS_BYTES);
    int*     sIdx = (int*)(smem + CB_LDS_BYTES + HN_LDS_BYTES);

    const int tid = threadIdx.x;

    // Stage f16 codebook into LDS with padded stride (72 halves = 36 dwords:
    // gcd(36,64)=4 -> 16 consecutive code rows land in distinct banks).
    for (int c = tid; c < KCODES * 8; c += 256) {
        const int row  = c >> 3;
        const int part = c & 7;
        *(half8v*)(cbL + row * LDS_ROW + part * 8) =
            *(const half8v*)(CBH + (size_t)row * DDIM + part * 8);
    }
    for (int i = tid; i < KCODES; i += 256) hnL[i] = HN[i];
    __syncthreads();

    const int lane = tid & 31;
    const int wave = tid >> 5;
    const int col  = lane & 15;     // WMMA B/C column handled by this lane
    const int ksel = lane >> 4;     // which K-half of the operand layout
    const int tileBase = blockIdx.x * (16 * WAVES) + wave * 16;

    // --- A operands: 16x64 f32 tile -> two 16x32 f16 WMMA operands.
    // CDNA5 A layout: lanes 0-15 (row=lane) hold K {0..7,16..23} in VGPR0..7,
    // lanes 16-31 (row=lane-16) hold K {8..15,24..31}.  a1 = same + 32.
    const int c0 = ksel * 8;
    const float* xr = X + (size_t)(tileBase + col) * DDIM + c0;
    half16v a0, a1;
    #pragma unroll
    for (int i = 0; i < 8; ++i) {
        a0[i]     = (_Float16)xr[i];
        a0[i + 8] = (_Float16)xr[16 + i];
        a1[i]     = (_Float16)xr[32 + i];
        a1[i + 8] = (_Float16)xr[48 + i];
    }

    float best[8];
    int   bidx[8];
    #pragma unroll
    for (int r = 0; r < 8; ++r) { best[r] = -3.0e38f; bidx[r] = 0; }

    union HU { half16v v; half8v h[2]; };

    // --- Column-tile loop over the 1024 codes (64 tiles of 16 codes).
    // Unrolled x2: two independent WMMA accumulator chains so the scheduler
    // can fill the WMMA->VALU hazard slots (4 coexec) with the other tile's
    // loads/compares instead of v_nops.
    #pragma unroll 2
    for (int j = 0; j < KCODES / 16; ++j) {
        const int code = j * 16 + col;
        // B layout (32x16 f16): lane-half h, n=col holds K {8h..8h+7, 16+8h..16+8h+7} (matches A).
        const _Float16* cr = cbL + code * LDS_ROW + ksel * 8;
        HU b0, b1;
        b0.h[0] = *(const half8v*)(cr);
        b0.h[1] = *(const half8v*)(cr + 16);
        b1.h[0] = *(const half8v*)(cr + 32);
        b1.h[1] = *(const half8v*)(cr + 48);

        const float nhn = -hnL[code];
        float8v acc;
        #pragma unroll
        for (int r = 0; r < 8; ++r) acc[r] = nhn;

        acc = __builtin_amdgcn_wmma_f32_16x16x32_f16(false, a0, false, b0.v,
                                                     (short)0, acc, false, false);
        acc = __builtin_amdgcn_wmma_f32_16x16x32_f16(false, a1, false, b1.v,
                                                     (short)0, acc, false, false);

        #pragma unroll
        for (int r = 0; r < 8; ++r) {
            if (acc[r] > best[r]) { best[r] = acc[r]; bidx[r] = code; }
        }
    }

    // --- Cross-lane argmax within each 16-lane half (lanes share the row set,
    // each lane owns a disjoint column residue). Branchless; tie -> lower idx.
    #pragma unroll
    for (int r = 0; r < 8; ++r) {
        float bv = best[r];
        int   bi = bidx[r];
        #pragma unroll
        for (int m = 1; m <= 8; m <<= 1) {
            const float ov = __shfl_xor(bv, m, 16);
            const int   oi = __shfl_xor(bi, m, 16);
            const int take = (int)(ov > bv) | ((int)(ov == bv) & (int)(oi < bi));
            bv = take ? ov : bv;
            bi = take ? oi : bi;
        }
        bidx[r] = bi;
    }

    // Lane 0 -> rows 0..7, lane 16 -> rows 8..15 (C/D layout halves).
    if (col == 0) {
        const int rbase = ksel * 8;
        #pragma unroll
        for (int r = 0; r < 8; ++r) {
            sIdx[wave * 16 + rbase + r] = bidx[r];
            atomicAdd(&counts[bidx[r]], 1u);
        }
    }
    __syncthreads();

    // --- Gather fp32 codebook row and scatter to NCHW output.
    // n = tileBase + row ; out[b, d, h, w] with b = n>>12, hw = n&4095.
    const int myIdx = sIdx[wave * 16 + col];
    const int n  = tileBase + col;
    const int bb = n >> 12;
    const int hw = n & (HW - 1);
    float* obase = out + (size_t)bb * CHW + hw;
    const float* crow = CB + (size_t)myIdx * DDIM;
    #pragma unroll
    for (int it = 0; it < 32; ++it) {
        const int d = it * 2 + ksel;     // lanes 0-15: even d, 16-31: odd d
        obase[(size_t)d * HW] = crow[d]; // 16 consecutive hw per half-wave
    }
}

// ---------------------------------------------------------------------------
// Kernel 3: perplexity = exp(-sum(e*log(e+1e-10))), e = counts/N.
// ---------------------------------------------------------------------------
__global__ void __launch_bounds__(256) vq_perplexity(const unsigned* __restrict__ counts,
                                                     float* __restrict__ outp) {
    __shared__ float red[256];
    float s = 0.0f;
    for (int k = threadIdx.x; k < KCODES; k += 256) {
        const float e = (float)counts[k] * (1.0f / (float)NROWS);
        s += e * logf(e + 1e-10f);
    }
    red[threadIdx.x] = s;
    __syncthreads();
    #pragma unroll
    for (int off = 128; off > 0; off >>= 1) {
        if (threadIdx.x < off) red[threadIdx.x] += red[threadIdx.x + off];
        __syncthreads();
    }
    if (threadIdx.x == 0) *outp = expf(-red[0]);
}

// ---------------------------------------------------------------------------
extern "C" void kernel_launch(void* const* d_in, const int* in_sizes, int n_in,
                              void* d_out, int out_size, void* d_ws, size_t ws_size,
                              hipStream_t stream) {
    const float* X  = (const float*)d_in[0];   // inputs  [32,64,64,64] f32 (NCHW)
    const float* CB = (const float*)d_in[1];   // codebook [1024,64] f32
    // d_in[2] = beta: unused in eval-mode forward.

    _Float16* cbh    = (_Float16*)d_ws;
    float*    hn     = (float*)((char*)d_ws + WS_HN_OFF);
    unsigned* counts = (unsigned*)((char*)d_ws + WS_CNT_OFF);
    float*    out    = (float*)d_out;

    hipMemsetAsync(counts, 0, KCODES * sizeof(unsigned), stream);
    vq_prep<<<KCODES / WAVES, 256, 0, stream>>>(CB, cbh, hn);
    vq_main<<<NROWS / (16 * WAVES), 256, SMEM_BYTES, stream>>>(X, CB, cbh, hn,
                                                               counts, out);
    vq_perplexity<<<1, 256, 0, stream>>>(counts, out + (out_size - 1));
}
